// TransformerBlock_27728308863301
// MI455X (gfx1250) — matrix operand order repaired and checked by reading, not compiled, not run on hardware
//
#include <hip/hip_runtime.h>
#include <hip/hip_bf16.h>

typedef _Float16 v16h __attribute__((ext_vector_type(16)));
typedef _Float16 v8h  __attribute__((ext_vector_type(8)));
typedef _Float16 v2h  __attribute__((ext_vector_type(2)));
typedef float    v8f  __attribute__((ext_vector_type(8)));

// ---------------------------------------------------------------------------
// WMMA fragment helpers (CDNA5 v_wmma_f32_16x16x32_f16 layouts, wave32)
// A (16x32 f16): lanes 0-15 -> M=lane; kb = (lane&16)?8:0
//   halves 0..7  -> K = k0 + kb + h   (contiguous, 16B aligned)
//   halves 8..15 -> K = k0 + 16 + kb + h-8 (contiguous, 16B aligned)
// B (32x16 f16): lane -> col = lane&15; kb = (lane&16)?8:0; halves -> K as for A (kb+h, 16+kb+h-8).
//   Weights are pre-swizzled so each (kt,nt,lane) fragment is 32 contiguous B.
// C/D (16x16 f32): element r -> row = r + ((lane&16)?8:0), col = lane&15.
// ---------------------------------------------------------------------------
__device__ __forceinline__ v16h load_a_frag(const _Float16* panel, int m0, int k0, int lane) {
    const int row = m0 + (lane & 15);
    const int kb  = (lane & 16) ? 8 : 0;
    const _Float16* p = panel + row * 128 + k0 + kb;
    v8h lo = *(const v8h*)(p);
    v8h hi = *(const v8h*)(p + 16);
    return __builtin_shufflevector(lo, hi, 0, 1, 2, 3, 4, 5, 6, 7,
                                   8, 9, 10, 11, 12, 13, 14, 15);
}

// Swizzled weight layout: frag = (kt*8 + nt)*32 + lane ; 16 halves contiguous.
__device__ __forceinline__ v16h load_b_frag(const _Float16* wswz, int nt, int kt, int lane) {
    const _Float16* p = wswz + (((kt << 3) | nt) << 5 | lane) * 16;
    v8h lo = *(const v8h*)(p);
    v8h hi = *(const v8h*)(p + 8);
    return __builtin_shufflevector(lo, hi, 0, 1, 2, 3, 4, 5, 6, 7,
                                   8, 9, 10, 11, 12, 13, 14, 15);
}

// One wave computes its 16x128 output tile: 8 N-tiles x 4 K-tiles of WMMA.
// C starts as inline 0 (no accumulator splat); bias is fused into writeback.
__device__ __forceinline__ void gemm_stage(const v16h* a, const _Float16* W,
                                           int lane, v8f* out) {
#pragma unroll
    for (int nt = 0; nt < 8; ++nt) {
        v8f acc = {};
#pragma unroll
        for (int kt = 0; kt < 4; ++kt) {
            v16h bf = load_b_frag(W, nt, kt, lane);
            acc = __builtin_amdgcn_wmma_f32_16x16x32_f16(false, a[kt], false, bf,
                                                         (short)0, acc, false, false);
        }
        out[nt] = acc;
    }
}

// 32KB weight panel copy: CDNA5 async global->LDS (ASYNCcnt, no VGPR data).
// Each lane issues 8x b128; generic LDS pointer truncates to LDS byte offset.
__device__ __forceinline__ void copy_w_async(const _Float16* src, _Float16* W, int tid) {
    const uint4* s = (const uint4*)src + tid;
    uint4* d = (uint4*)W + tid;
#pragma unroll
    for (int i = 0; i < 8; ++i) {
        const unsigned lds = (unsigned)(unsigned long long)(d + i * 256);
        asm volatile("global_load_async_to_lds_b128 %0, %1, off"
                     :: "v"(lds), "v"((const void*)(s + i * 256)) : "memory");
    }
    asm volatile("s_wait_asynccnt 0" ::: "memory");
}

__device__ __forceinline__ void load_w(const _Float16* src, _Float16* W, int tid) {
    __syncthreads();
    copy_w_async(src, W, tid);
    __syncthreads();
}

// ---------------------------------------------------------------------------
// Kernel 1: fp32 -> f16 weight conversion + B-fragment swizzle.
// order: 0=W0b 1=W1 2=Wq 3=Wk 4=Wv 5=Wd2 6=Wg1 7=Wg2 8=W2
// ---------------------------------------------------------------------------
__global__ __launch_bounds__(256) void convert_weights_kernel(
    const float* w0b, const float* w1, const float* wq, const float* wk,
    const float* wv, const float* wd2, const float* wg1, const float* wg2,
    const float* w2, _Float16* dst) {
    const int i = blockIdx.x * 256 + threadIdx.x;   // 0 .. 147455
    const int m = i >> 14, e = i & 16383;
    const int k = e >> 7, n = e & 127;
    const int kt = k >> 5, kb16 = (k >> 3) & 1, h = (k & 7) | (((k >> 4) & 1) << 3);
    const int nt = n >> 4, lane = (kb16 << 4) | (n & 15);
    const int dstIdx = ((((kt << 3) | nt) << 5) | lane) * 16 + h;
    const float* src[9] = {w0b, w1, wq, wk, wv, wd2, wg1, wg2, w2};
    dst[m * 16384 + dstIdx] = (_Float16)src[m][e];
}

// ---------------------------------------------------------------------------
// Kernel 2: KNN. One thread per query point, candidates tiled through LDS,
// top-16 kept sorted (ascending distance, ties keep lower index -> top_k order).
// ---------------------------------------------------------------------------
__global__ __launch_bounds__(256) void knn_kernel(const float* xyz, int* knnOut) {
    __shared__ float cs[256 * 3];
    const int b = blockIdx.y;
    const int n = blockIdx.x * 256 + threadIdx.x;
    const float* base = xyz + b * 4096 * 3;
    const float qx = base[n * 3 + 0], qy = base[n * 3 + 1], qz = base[n * 3 + 2];
    float dist[16];
    int   idxr[16];
#pragma unroll
    for (int i = 0; i < 16; ++i) { dist[i] = 3.4e38f; idxr[i] = 0; }
    for (int t = 0; t < 4096; t += 256) {
        __syncthreads();
        for (int i = threadIdx.x; i < 768; i += 256) cs[i] = base[t * 3 + i];
        __syncthreads();
        for (int jj = 0; jj < 256; ++jj) {
            const float dx = qx - cs[jj * 3 + 0];
            const float dy = qy - cs[jj * 3 + 1];
            const float dz = qz - cs[jj * 3 + 2];
            const float d = dx * dx + dy * dy + dz * dz;
            if (d < dist[15]) {
                dist[15] = d; idxr[15] = t + jj;
#pragma unroll
                for (int s = 15; s > 0; --s) {
                    if (dist[s] < dist[s - 1]) {
                        float td = dist[s]; dist[s] = dist[s - 1]; dist[s - 1] = td;
                        int ti = idxr[s]; idxr[s] = idxr[s - 1]; idxr[s - 1] = ti;
                    }
                }
            }
        }
    }
#pragma unroll
    for (int i = 0; i < 16; ++i) knnOut[(b * 4096 + n) * 16 + i] = idxr[i];
}

// ---------------------------------------------------------------------------
// Kernel 3: per-point MLP chain. Block = 128 rows, 8 waves, wave owns 16 rows.
// pre = relu(xyz@W0a+b0a)@W0b + b0b ; x = pre@W1 + b1 ; q/k/v = x@Wq/Wk/Wv.
// ---------------------------------------------------------------------------
__global__ __launch_bounds__(256) void point_mlp_kernel(
    const float* xyz, const float* W0a, const float* b0a,
    const float* b0bP, const float* b1P, const _Float16* wf16,
    float* preO, float* qO, float* xkO, float* xvO) {
    __shared__ __align__(16) _Float16 A[16384];
    __shared__ __align__(16) _Float16 W[16384];
    __shared__ float xyzs[384];
    __shared__ float b0bs[128], b1s[128];
    const int tid = threadIdx.x, lane = tid & 31, wave = tid >> 5;
    const int rowBase = blockIdx.x * 128;

    for (int i = tid; i < 384; i += 256) xyzs[i] = xyz[rowBase * 3 + i];
    if (tid < 128) { b0bs[tid] = b0bP[tid]; b1s[tid] = b1P[tid]; }
    __syncthreads();
    for (int i = tid; i < 8192; i += 256) {        // pairs of columns
        const int r = i >> 6, c = (i & 63) * 2;
        const float x0 = xyzs[r * 3 + 0], x1 = xyzs[r * 3 + 1], x2 = xyzs[r * 3 + 2];
        float va = x0 * W0a[c] + x1 * W0a[128 + c] + x2 * W0a[256 + c] + b0a[c];
        float vb = x0 * W0a[c + 1] + x1 * W0a[129 + c] + x2 * W0a[257 + c] + b0a[c + 1];
        v2h p; p[0] = (_Float16)fmaxf(va, 0.f); p[1] = (_Float16)fmaxf(vb, 0.f);
        *(v2h*)&A[r * 128 + c] = p;
    }

    const int m0 = wave * 16, colb = lane & 15, rh = (lane & 16) ? 8 : 0;
    v16h a[4];
    v8f  out[8];

    // ---- stage 0: pre = hid0 @ W0b + b0b  (store fp32, keep f16 in A)
    load_w(wf16 + 0 * 16384, W, tid);
#pragma unroll
    for (int kt = 0; kt < 4; ++kt) a[kt] = load_a_frag(A, m0, kt * 32, lane);
    gemm_stage(a, W, lane, out);
#pragma unroll
    for (int nt = 0; nt < 8; ++nt) {
        const int col = nt * 16 + colb;
        const float bv = b0bs[col];
#pragma unroll
        for (int r = 0; r < 8; ++r) {
            const int row = m0 + r + rh;
            const float v = out[nt][r] + bv;
            preO[(rowBase + row) * 128 + col] = v;
            A[row * 128 + col] = (_Float16)v;
        }
    }

    // ---- stage 1: x = pre @ W1 + b1  (keep f16 in A)
    load_w(wf16 + 1 * 16384, W, tid);
#pragma unroll
    for (int kt = 0; kt < 4; ++kt) a[kt] = load_a_frag(A, m0, kt * 32, lane);
    gemm_stage(a, W, lane, out);
#pragma unroll
    for (int nt = 0; nt < 8; ++nt) {
        const int col = nt * 16 + colb;
        const float bv = b1s[col];
#pragma unroll
        for (int r = 0; r < 8; ++r)
            A[(m0 + r + rh) * 128 + col] = (_Float16)(out[nt][r] + bv);
    }

    // ---- stages 2..4: q/k/v = x @ {Wq,Wk,Wv} (no bias) — x frags reused
    load_w(wf16 + 2 * 16384, W, tid);
#pragma unroll
    for (int kt = 0; kt < 4; ++kt) a[kt] = load_a_frag(A, m0, kt * 32, lane);
    gemm_stage(a, W, lane, out);
#pragma unroll
    for (int nt = 0; nt < 8; ++nt)
#pragma unroll
        for (int r = 0; r < 8; ++r)
            qO[(rowBase + m0 + r + rh) * 128 + nt * 16 + colb] = out[nt][r];

    load_w(wf16 + 3 * 16384, W, tid);
    gemm_stage(a, W, lane, out);
#pragma unroll
    for (int nt = 0; nt < 8; ++nt)
#pragma unroll
        for (int r = 0; r < 8; ++r)
            xkO[(rowBase + m0 + r + rh) * 128 + nt * 16 + colb] = out[nt][r];

    load_w(wf16 + 4 * 16384, W, tid);
    gemm_stage(a, W, lane, out);
#pragma unroll
    for (int nt = 0; nt < 8; ++nt)
#pragma unroll
        for (int r = 0; r < 8; ++r)
            xvO[(rowBase + m0 + r + rh) * 128 + nt * 16 + colb] = out[nt][r];
}

// ---------------------------------------------------------------------------
// Kernel 4: neighbor-level MLPs + softmax + weighted sum.
// Block = 8 query points (128 neighbor rows); wave w owns query w (16 rows),
// so softmax over K=16 lives in its own M-tile: in-register + shfl_xor(16).
// ---------------------------------------------------------------------------
__global__ __launch_bounds__(256) void attn_kernel(
    const float* xyz, const int* knn,
    const float* Wd1, const float* bd1P, const float* bd2P,
    const float* bg1P, const float* bg2P,
    const _Float16* wf16, const float* qG, const float* xkG, const float* xvG,
    float* attnOut, float* resAcc) {
    __shared__ __align__(16) _Float16 W[16384];
    __shared__ __align__(16) _Float16 A[16384];
    __shared__ __align__(16) _Float16 POS[16384];
    __shared__ __align__(16) _Float16 VF[16384];
    __shared__ float qs[1024];
    __shared__ float wd1s[384];
    __shared__ float bd1s[128], bd2s[128], bg1s[128], bg2s[128];
    __shared__ float deltas[384];
    __shared__ int idxs[128];

    const int tid = threadIdx.x, lane = tid & 31, wave = tid >> 5;
    const int pBase = blockIdx.x * 8;      // global point row in [0, B*N)
    const int b = pBase >> 12;             // N = 4096
    const float* xyzB = xyz + b * 4096 * 3;

    copy_w_async(wf16 + 5 * 16384, W, tid);      // Wd2
    for (int i = tid; i < 384; i += 256) wd1s[i] = Wd1[i];
    for (int i = tid; i < 128; i += 256) {
        bd1s[i] = bd1P[i]; bd2s[i] = bd2P[i]; bg1s[i] = bg1P[i]; bg2s[i] = bg2P[i];
        idxs[i] = knn[pBase * 16 + i];
    }
    for (int i = tid; i < 1024; i += 256) qs[i] = qG[pBase * 128 + i];
    __syncthreads();
    if (tid < 128) {
        const int r = tid, qp = pBase + (r >> 4), j = idxs[r];
        deltas[r * 3 + 0] = xyz[qp * 3 + 0] - xyzB[j * 3 + 0];
        deltas[r * 3 + 1] = xyz[qp * 3 + 1] - xyzB[j * 3 + 1];
        deltas[r * 3 + 2] = xyz[qp * 3 + 2] - xyzB[j * 3 + 2];
    }
    __syncthreads();
    for (int i = tid; i < 8192; i += 256) {        // pairs of columns
        const int r = i >> 6, c = (i & 63) * 2;
        const float d0 = deltas[r * 3 + 0], d1 = deltas[r * 3 + 1], d2 = deltas[r * 3 + 2];
        float va = d0 * wd1s[c] + d1 * wd1s[128 + c] + d2 * wd1s[256 + c] + bd1s[c];
        float vb = d0 * wd1s[c + 1] + d1 * wd1s[129 + c] + d2 * wd1s[257 + c] + bd1s[c + 1];
        v2h p; p[0] = (_Float16)fmaxf(va, 0.f); p[1] = (_Float16)fmaxf(vb, 0.f);
        *(v2h*)&A[r * 128 + c] = p;
        const float2 xv2 = *(const float2*)&xvG[(b * 4096 + idxs[r]) * 128 + c];
        v2h vv; vv[0] = (_Float16)xv2.x; vv[1] = (_Float16)xv2.y;
        *(v2h*)&VF[r * 128 + c] = vv;
    }
    __syncthreads();

    const int m0 = wave * 16, colb = lane & 15, rh = (lane & 16) ? 8 : 0;
    v16h a[4];
    v8f  out[8];

    // ---- pos = relu(delta@Wd1+bd1) @ Wd2 + bd2 ; then h = q - kf + pos -> A
    __builtin_prefetch(wf16 + 6 * 16384, 0, 1);
#pragma unroll
    for (int kt = 0; kt < 4; ++kt) a[kt] = load_a_frag(A, m0, kt * 32, lane);
    gemm_stage(a, W, lane, out);
#pragma unroll
    for (int nt = 0; nt < 8; ++nt) {
        const int col = nt * 16 + colb;
        const float bv = bd2s[col];
        const float qv = qs[wave * 128 + col];
#pragma unroll
        for (int r = 0; r < 8; ++r) {
            const int row = m0 + r + rh;
            const float p = out[nt][r] + bv;
            POS[row * 128 + col] = (_Float16)p;
            const int j = b * 4096 + idxs[row];
            const float hv = qv - xkG[j * 128 + col] + p;
            A[row * 128 + col] = (_Float16)hv;
        }
    }
    load_w(wf16 + 6 * 16384, W, tid);            // Wg1

    // ---- g1 = relu(h @ Wg1 + bg1) -> A
    __builtin_prefetch(wf16 + 7 * 16384, 0, 1);
#pragma unroll
    for (int kt = 0; kt < 4; ++kt) a[kt] = load_a_frag(A, m0, kt * 32, lane);
    gemm_stage(a, W, lane, out);
#pragma unroll
    for (int nt = 0; nt < 8; ++nt) {
        const int col = nt * 16 + colb;
        const float bv = bg1s[col];
#pragma unroll
        for (int r = 0; r < 8; ++r)
            A[(m0 + r + rh) * 128 + col] = (_Float16)fmaxf(out[nt][r] + bv, 0.f);
    }
    load_w(wf16 + 7 * 16384, W, tid);            // Wg2

    // ---- logits = (g1 @ Wg2 + bg2)/sqrt(D) ; softmax over K ; weighted sum
#pragma unroll
    for (int kt = 0; kt < 4; ++kt) a[kt] = load_a_frag(A, m0, kt * 32, lane);
    gemm_stage(a, W, lane, out);
    const float scale = 0.0883883476483184f;   // 1/sqrt(128)
#pragma unroll
    for (int nt = 0; nt < 8; ++nt) {
        const int col = nt * 16 + colb;
        const float bv = bg2s[col];
        float mx = -3.4e38f;
#pragma unroll
        for (int r = 0; r < 8; ++r) {
            out[nt][r] = (out[nt][r] + bv) * scale;
            mx = fmaxf(mx, out[nt][r]);
        }
        mx = fmaxf(mx, __shfl_xor(mx, 16));
        float e[8], sum = 0.f;
#pragma unroll
        for (int r = 0; r < 8; ++r) { e[r] = __expf(out[nt][r] - mx); sum += e[r]; }
        sum += __shfl_xor(sum, 16);
        const float inv = 1.f / sum;
        float racc = 0.f;
#pragma unroll
        for (int r = 0; r < 8; ++r) {
            const int k = r + rh, row = m0 + k;
            const float at = e[r] * inv;
            attnOut[((long)(pBase + wave) * 16 + k) * 128 + col] = at;
            racc += at * ((float)VF[row * 128 + col] + (float)POS[row * 128 + col]);
        }
        racc += __shfl_xor(racc, 16);
        if (lane < 16) resAcc[(pBase + wave) * 128 + col] = racc;
    }
}

// ---------------------------------------------------------------------------
// Kernel 5: res = resAcc @ W2 + b2 + pre
// ---------------------------------------------------------------------------
__global__ __launch_bounds__(256) void final_kernel(
    const float* resAccP, const float* pre, const float* b2P,
    const _Float16* w2f16, float* outRes) {
    __shared__ __align__(16) _Float16 A[16384];
    __shared__ __align__(16) _Float16 W[16384];
    __shared__ float bias[128];
    const int tid = threadIdx.x, lane = tid & 31, wave = tid >> 5;
    const int rowBase = blockIdx.x * 128;
    for (int i = tid; i < 8192; i += 256) {
        const float2 v = *(const float2*)&resAccP[rowBase * 128 + i * 2];
        v2h p; p[0] = (_Float16)v.x; p[1] = (_Float16)v.y;
        *(v2h*)&A[i * 2] = p;
    }
    copy_w_async(w2f16, W, tid);
    if (tid < 128) bias[tid] = b2P[tid];
    __syncthreads();
    const int m0 = wave * 16, colb = lane & 15, rh = (lane & 16) ? 8 : 0;
    v16h a[4];
    v8f  out[8];
#pragma unroll
    for (int kt = 0; kt < 4; ++kt) a[kt] = load_a_frag(A, m0, kt * 32, lane);
    gemm_stage(a, W, lane, out);
#pragma unroll
    for (int nt = 0; nt < 8; ++nt) {
        const int col = nt * 16 + colb;
        const float bv = bias[col];
#pragma unroll
        for (int r = 0; r < 8; ++r) {
            const int row = rowBase + m0 + r + rh;
            outRes[row * 128 + col] = out[nt][r] + bv + pre[row * 128 + col];
        }
    }
}

// ---------------------------------------------------------------------------
extern "C" void kernel_launch(void* const* d_in, const int* in_sizes, int n_in,
                              void* d_out, int out_size, void* d_ws, size_t ws_size,
                              hipStream_t stream) {
    const float* xyz = (const float*)d_in[0];
    const float* W0a = (const float*)d_in[1];
    const float* b0a = (const float*)d_in[2];
    const float* W0b = (const float*)d_in[3];
    const float* b0b = (const float*)d_in[4];
    const float* W1  = (const float*)d_in[5];
    const float* b1  = (const float*)d_in[6];
    const float* W2  = (const float*)d_in[7];
    const float* b2  = (const float*)d_in[8];
    const float* Wd1 = (const float*)d_in[9];
    const float* bd1 = (const float*)d_in[10];
    const float* Wd2 = (const float*)d_in[11];
    const float* bd2 = (const float*)d_in[12];
    const float* Wg1 = (const float*)d_in[13];
    const float* bg1 = (const float*)d_in[14];
    const float* Wg2 = (const float*)d_in[15];
    const float* bg2 = (const float*)d_in[16];
    const float* Wq  = (const float*)d_in[17];
    const float* Wk  = (const float*)d_in[18];
    const float* Wv  = (const float*)d_in[19];

    char* ws = (char*)d_ws;
    _Float16* wf16 = (_Float16*)(ws + 0);          // 9*16384*2      = 294912 B
    int*   knnW    = (int*)(ws + 294912);          // 262144*4       -> 1343488
    float* preW    = (float*)(ws + 1343488);       // 16384*128*4    -> 9732096
    float* qW      = (float*)(ws + 9732096);       //                -> 18120704
    float* xkW     = (float*)(ws + 18120704);      //                -> 26509312
    float* xvW     = (float*)(ws + 26509312);      //                -> 34897920
    float* resW    = (float*)(ws + 34897920);      //                -> 43286528

    float* outRes  = (float*)d_out;                // [4,4096,128]
    float* outAttn = outRes + 4 * 4096 * 128;      // [4,4096,16,128]

    convert_weights_kernel<<<576, 256, 0, stream>>>(W0b, W1, Wq, Wk, Wv, Wd2, Wg1,
                                                    Wg2, W2, wf16);
    knn_kernel<<<dim3(16, 4), 256, 0, stream>>>(xyz, knnW);
    point_mlp_kernel<<<128, 256, 0, stream>>>(xyz, W0a, b0a, b0b, b1, wf16,
                                              preW, qW, xkW, xvW);
    attn_kernel<<<2048, 256, 0, stream>>>(xyz, knnW, Wd1, bd1, bd2, bg1, bg2, wf16,
                                          qW, xkW, xvW, outAttn, resW);
    final_kernel<<<128, 256, 0, stream>>>(resW, preW, b2, wf16 + 8 * 16384, outRes);
}
